// VisualGraph_2439541424477
// MI455X (gfx1250) — matrix operand order repaired and checked by reading, not compiled, not run on hardware
//
#include <hip/hip_runtime.h>

// ---------------------------------------------------------------------------
// Sizes from the reference
// ---------------------------------------------------------------------------
#define N_IMAGE   128
#define N_CAPTION 128
#define K_REGION  36
#define N_WORD    48
#define EMBED     1024
#define NUM_BLOCK 32
#define N_BLOCK   32
#define N_KERNELS 8
#define HID_DIM   128
#define LAMBDA_SM 20.0f
#define EPS_F     1e-8f

typedef __attribute__((ext_vector_type(16))) __bf16 v16bf;
typedef __attribute__((ext_vector_type(8)))  __bf16 v8bf;
typedef __attribute__((ext_vector_type(8)))  float  v8f;

union fragU { v16bf v; v8bf h[2]; };

// D = A(16x32 bf16) * B(32x16 bf16) + C(16x16 f32)
static __device__ __forceinline__ v8f wmma_bf16(v16bf a, v16bf b, v8f c) {
  return __builtin_amdgcn_wmma_f32_16x16x32_bf16(false, a, false, b,
                                                 (short)0, c, false, false);
}

// A-matrix fragment (16x32, row-major source: row m, K contiguous).
// Lane L<16: row m0+L, K chunks [k0+0..7] and [k0+16..23].
// Lane L>=16: row m0+L-16, K chunks [k0+8..15] and [k0+24..31].
static __device__ __forceinline__ v16bf load_a(const __bf16* base, int stride,
                                               int m0, int k0, int lane) {
  int r = m0 + (lane & 15);
  int k = k0 + ((lane & 16) ? 8 : 0);
  const __bf16* p = base + (size_t)r * stride + k;
  fragU f;
  f.h[0] = *(const v8bf*)p;
  f.h[1] = *(const v8bf*)(p + 16);
  return f.v;
}

// B-matrix fragment (32x16). Source is "B-transposed" row-major:
// srcT[n][k] = B[k][n]. Lane L: column n0+(L&15), K chunks [k0+0..7] and
// [k0+16..23] (lanes 0-15) or [k0+8..15] and [k0+24..31] (lanes 16-31).
static __device__ __forceinline__ v16bf load_b(const __bf16* baseT, int stride,
                                               int n0, int k0, int lane) {
  int r = n0 + (lane & 15);
  int k = k0 + ((lane & 16) ? 8 : 0);
  const __bf16* p = baseT + (size_t)r * stride + k;
  fragU f;
  f.h[0] = *(const v8bf*)p;
  f.h[1] = *(const v8bf*)(p + 16);
  return f.v;
}

// D/C layout: VGPR v, lane L -> row m0 + v + 8*(L>=16), col n0 + (L&15)
static __device__ __forceinline__ void store_d_f32(float* dst, int stride,
                                                   int m0, int n0, int lane,
                                                   v8f a) {
  int n = n0 + (lane & 15);
  int m = m0 + ((lane & 16) ? 8 : 0);
#pragma unroll
  for (int v = 0; v < 8; ++v) dst[(m + v) * stride + n] = a[v];
}

// ---------------------------------------------------------------------------
// Setup kernels
// ---------------------------------------------------------------------------
__global__ void cvt_img_kernel(const float* __restrict__ img,
                               __bf16* __restrict__ dst) {
  int idx = blockIdx.x * 256 + threadIdx.x;      // 128*48*1024 threads
  int d = idx & 1023;
  int r = (idx >> 10) % 48;
  int b = idx / (48 * 1024);
  float v = 0.f;
  if (r < K_REGION) v = img[((size_t)b * K_REGION + r) * EMBED + d];
  dst[idx] = (__bf16)v;
}

__global__ void cvt_cap_kernel(const float* __restrict__ cap,
                               __bf16* __restrict__ dst) {
  int idx = blockIdx.x * 256 + threadIdx.x;      // 128*48*1024 threads
  dst[idx] = (__bf16)cap[idx];
}

__global__ void cvt_capT_kernel(const float* __restrict__ cap,
                                __bf16* __restrict__ dst) {
  int idx = blockIdx.x * 256 + threadIdx.x;      // 128*1024*64 threads
  int l = idx & 63;
  int d = (idx >> 6) & 1023;
  int c = idx >> 16;
  float v = 0.f;
  if (l < N_WORD) v = cap[((size_t)c * N_WORD + l) * EMBED + d];
  dst[idx] = (__bf16)v;
}

__global__ void qnorm_kernel(const float* __restrict__ img,
                             float* __restrict__ qn) {
  int idx = blockIdx.x * 256 + threadIdx.x;      // 128*36*32 threads
  int blk = idx & 31;
  int r = (idx >> 5) % K_REGION;
  int b = idx / (K_REGION * 32);
  const float* q = img + ((size_t)b * K_REGION + r) * EMBED + blk * N_BLOCK;
  float s = 0.f;
#pragma unroll
  for (int d = 0; d < N_BLOCK; ++d) s += q[d] * q[d];
  qn[idx] = fmaxf(sqrtf(s), EPS_F);
}

// sgw[b,i,k] = sum_j gw[b,i,j,k]  (gw normalized over k per (b,i,j))
__global__ void sgw_kernel(const float* __restrict__ bbox,
                           const float* __restrict__ mr,
                           const float* __restrict__ pr,
                           const float* __restrict__ mt,
                           const float* __restrict__ pt,
                           float* __restrict__ sgw) {
  int idx = blockIdx.x * 256 + threadIdx.x;
  if (idx >= N_IMAGE * K_REGION) return;
  int b = idx / K_REGION, i = idx % K_REGION;
  const float* bi = bbox + ((size_t)b * K_REGION + i) * 4;
  float cxi = 0.5f * (bi[0] + bi[2]);
  float cyi = 0.5f * (bi[1] + bi[3]);
  float m_r[8], p_r[8], m_t[8], p_t[8], acc[8];
#pragma unroll
  for (int k = 0; k < 8; ++k) {
    m_r[k] = mr[k]; p_r[k] = pr[k]; m_t[k] = mt[k]; p_t[k] = pt[k];
    acc[k] = 0.f;
  }
  for (int j = 0; j < K_REGION; ++j) {
    const float* bj = bbox + ((size_t)b * K_REGION + j) * 4;
    float dx = cxi - 0.5f * (bj[0] + bj[2]);
    float dy = cyi - 0.5f * (bj[1] + bj[3]);
    float rho = sqrtf(dx * dx + dy * dy);
    float theta = atan2f(dx, dy);
    float w[8], s = 0.f;
#pragma unroll
    for (int k = 0; k < 8; ++k) {
      float dr = rho - m_r[k];
      float wr = expf(-0.5f * dr * dr / (1e-14f + p_r[k] * p_r[k]));
      float a1 = fabsf(theta - m_t[k]);
      float a2 = fabsf(6.28318530717958647f - a1);
      float am = fminf(a1, a2);
      float wt = expf(-0.5f * am * am / (1e-14f + p_t[k] * p_t[k]));
      float ww = wr * wt;
      if (ww != ww) ww = 0.f;     // nan -> 0 as in reference
      w[k] = ww; s += ww;
    }
    float inv = 1.f / s;          // matches reference (nan if s==0)
#pragma unroll
    for (int k = 0; k < 8; ++k) acc[k] += w[k] * inv;
  }
#pragma unroll
  for (int k = 0; k < 8; ++k) sgw[(size_t)idx * 8 + k] = acc[k];
}

// weight-normalized W1 (bf16, row-major 128x128), w2 (f32, 128),
// conv_wT[co=k*16+o][f] (bf16, 128x32)
__global__ void prep_weights_kernel(const float* __restrict__ o1v,
                                    const float* __restrict__ o1g,
                                    const float* __restrict__ o2v,
                                    const float* __restrict__ o2g,
                                    const float* __restrict__ conv_w,
                                    __bf16* __restrict__ w1n,
                                    float* __restrict__ w2n,
                                    __bf16* __restrict__ convT) {
  int n = threadIdx.x;            // 128 threads
  float s = 0.f;
  for (int k = 0; k < HID_DIM; ++k) {
    float v = o1v[n * HID_DIM + k];
    s += v * v;
  }
  float sc = o1g[n] / sqrtf(s);
  for (int k = 0; k < HID_DIM; ++k)
    w1n[n * HID_DIM + k] = (__bf16)(o1v[n * HID_DIM + k] * sc);

  float s2 = 0.f;
  for (int k = 0; k < HID_DIM; ++k) s2 += o2v[k] * o2v[k];
  w2n[n] = o2v[n] * (o2g[0] / sqrtf(s2));

  // conv_w shape (8, 32, 16) = [k][f][o] -> convT[(k*16+o)*32 + f]
  for (int t = n; t < 8 * 32 * 16; t += 128) {
    int k = t >> 9;
    int rem = t & 511;
    int o = rem >> 5;
    int f = rem & 31;
    convT[t] = (__bf16)conv_w[k * 512 + f * 16 + o];
  }
}

// ---------------------------------------------------------------------------
// Main kernel: one wave32 per (image b, caption c) pair
// ---------------------------------------------------------------------------
__global__ __launch_bounds__(32) void pair_kernel(
    const float* __restrict__ images,       // f32 (128,36,1024)
    const __bf16* __restrict__ img_bf,      // (128,48,1024) padded
    const __bf16* __restrict__ cap_bf,      // (128,48,1024)
    const __bf16* __restrict__ capT_bf,     // (128,1024,64) padded l
    const float* __restrict__ qnorm,        // (128,36,32)
    const float* __restrict__ sgw,          // (128,36,8)
    const __bf16* __restrict__ convT,       // (128,32)
    const __bf16* __restrict__ w1n,         // (128,128)
    const float* __restrict__ b1,           // (128)
    const float* __restrict__ w2n,          // (128)
    const float* __restrict__ b2,           // (1)
    float* __restrict__ out)                // (128,128) [b][c]
{
  const int b = blockIdx.x;
  const int c = blockIdx.y;
  const int lane = threadIdx.x;

  // attn / ctx / h have disjoint lifetimes -> union them (12 KB instead of
  // 27 KB; total static LDS 21.5 KB -> ~14 resident single-wave workgroups
  // per 320 KB WGP for latency hiding).
  __shared__ union SharedU {
    float  attn[48][48];   // phase 1-2: attn[l][r]
    float  ctx[48][32];    // phase 3:   ctx tile (r x 32 d)
    __bf16 h[48][128];     // phase 4-5: h row-major, M-padded
  } u;
  __shared__ __bf16 attnT_s[48][64];  // softmaxed, transposed, K-padded
  __shared__ __bf16 mv_s[48][32];     // mv row-major, M-padded

  // ---- GEMM 1: attn[l][r] = sum_d cap[l,d] * img[r,d] -------------------
  {
    const __bf16* A = cap_bf + (size_t)c * 48 * EMBED;
    const __bf16* B = img_bf + (size_t)b * 48 * EMBED;
    v8f zero = {0.f, 0.f, 0.f, 0.f, 0.f, 0.f, 0.f, 0.f};
    v8f acc[3][3];
#pragma unroll
    for (int m = 0; m < 3; ++m)
#pragma unroll
      for (int n = 0; n < 3; ++n) acc[m][n] = zero;

    // per-lane base rows for prefetch addresses
    const __bf16* pa = A + (size_t)(lane & 15) * EMBED;
    const __bf16* pb = B + (size_t)(lane & 15) * EMBED;

    for (int k = 0; k < EMBED; k += 32) {
      // unconditional near-cache prefetch two k-tiles ahead; branchless
      // scalar clamp keeps the loop a single basic block
      int kp = (k + 64 < EMBED) ? (k + 64) : k;
#pragma unroll
      for (int m = 0; m < 3; ++m)
        __builtin_prefetch(pa + (size_t)(16 * m) * EMBED + kp, 0, 3);
#pragma unroll
      for (int n = 0; n < 3; ++n)
        __builtin_prefetch(pb + (size_t)(16 * n) * EMBED + kp, 0, 3);

      v16bf af[3], bf[3];
#pragma unroll
      for (int m = 0; m < 3; ++m) af[m] = load_a(A, EMBED, 16 * m, k, lane);
#pragma unroll
      for (int n = 0; n < 3; ++n) bf[n] = load_b(B, EMBED, 16 * n, k, lane);
#pragma unroll
      for (int m = 0; m < 3; ++m)
#pragma unroll
        for (int n = 0; n < 3; ++n)
          acc[m][n] = wmma_bf16(af[m], bf[n], acc[m][n]);
    }
#pragma unroll
    for (int m = 0; m < 3; ++m)
#pragma unroll
      for (int n = 0; n < 3; ++n)
        store_d_f32(&u.attn[0][0], 48, 16 * m, 16 * n, lane, acc[m][n]);
  }
  __syncthreads();

  // ---- leaky relu + l2norm over r (per row l) ---------------------------
  for (int l = lane; l < 48; l += 32) {
    float ss = 0.f;
    for (int r = 0; r < 48; ++r) {
      float x = u.attn[l][r];
      x = (x > 0.f) ? x : 0.1f * x;
      u.attn[l][r] = x;
      ss += x * x;
    }
    float inv = 1.f / (sqrtf(ss) + EPS_F);
    for (int r = 0; r < 48; ++r) u.attn[l][r] *= inv;
  }
  __syncthreads();

  // ---- transpose + softmax over l (per row r), write bf16 ---------------
  for (int r = lane; r < 48; r += 32) {
    float mx = -3.4e38f;
    for (int l = 0; l < 48; ++l) mx = fmaxf(mx, u.attn[l][r] * LAMBDA_SM);
    float sum = 0.f;
    for (int l = 0; l < 48; ++l) sum += expf(u.attn[l][r] * LAMBDA_SM - mx);
    float inv = 1.f / sum;
    for (int l = 0; l < 48; ++l)
      attnT_s[r][l] = (__bf16)(expf(u.attn[l][r] * LAMBDA_SM - mx) * inv);
    for (int l = 48; l < 64; ++l) attnT_s[r][l] = (__bf16)0.f;
  }
  // zero mv (covers padded rows 36..47)
  for (int i = lane; i < 48 * 32; i += 32) (&mv_s[0][0])[i] = (__bf16)0.f;
  __syncthreads();   // attn region is dead from here on -> reuse as ctx

  // ---- GEMM 2 + block cosine: ctx = attnT @ cap, mv[r][blk] -------------
  {
    const __bf16* CT = capT_bf + (size_t)c * EMBED * 64;
    v8f zero = {0.f, 0.f, 0.f, 0.f, 0.f, 0.f, 0.f, 0.f};
    for (int blk = 0; blk < NUM_BLOCK; ++blk) {
      v8f cacc[3][2];
#pragma unroll
      for (int m = 0; m < 3; ++m) { cacc[m][0] = zero; cacc[m][1] = zero; }
#pragma unroll
      for (int ks = 0; ks < 64; ks += 32) {
        v16bf af[3];
#pragma unroll
        for (int m = 0; m < 3; ++m)
          af[m] = load_a(&attnT_s[0][0], 64, 16 * m, ks, lane);
        v16bf bf0 = load_b(CT, 64, blk * 32, ks, lane);
        v16bf bf1 = load_b(CT, 64, blk * 32 + 16, ks, lane);
#pragma unroll
        for (int m = 0; m < 3; ++m) {
          cacc[m][0] = wmma_bf16(af[m], bf0, cacc[m][0]);
          cacc[m][1] = wmma_bf16(af[m], bf1, cacc[m][1]);
        }
      }
#pragma unroll
      for (int m = 0; m < 3; ++m) {
        store_d_f32(&u.ctx[0][0], 32, 16 * m, 0, lane, cacc[m][0]);
        store_d_f32(&u.ctx[0][0], 32, 16 * m, 16, lane, cacc[m][1]);
      }
      __syncthreads();
      for (int r = lane; r < K_REGION; r += 32) {
        float num = 0.f, cn = 0.f;
        const float* q =
            images + ((size_t)b * K_REGION + r) * EMBED + blk * N_BLOCK;
        for (int d = 0; d < N_BLOCK; ++d) {
          float cv = u.ctx[r][d];
          num += q[d] * cv;
          cn += cv * cv;
        }
        float den = qnorm[((size_t)b * K_REGION + r) * NUM_BLOCK + blk] *
                    fmaxf(sqrtf(cn), EPS_F);
        mv_s[r][blk] = (__bf16)(num / den);
      }
      __syncthreads();
    }
  }

  // ---- conv GEMM: p = mv @ W(32x128), h = sgw-scale, store bf16 ---------
  // (ctx region is dead -> reuse union as h)
  {
    const float* SG = sgw + (size_t)b * K_REGION * N_KERNELS;
    v8f zero = {0.f, 0.f, 0.f, 0.f, 0.f, 0.f, 0.f, 0.f};
#pragma unroll
    for (int nt = 0; nt < 8; ++nt) {       // output kernel k == nt
      v16bf bfc = load_b(convT, 32, nt * 16, 0, lane);
      v8f p[3];
#pragma unroll
      for (int m = 0; m < 3; ++m) {
        v16bf afm = load_a(&mv_s[0][0], 32, 16 * m, 0, lane);
        p[m] = wmma_bf16(afm, bfc, zero);
      }
      int n = nt * 16 + (lane & 15);
#pragma unroll
      for (int m = 0; m < 3; ++m) {
        int mbase = m * 16 + ((lane & 16) ? 8 : 0);
#pragma unroll
        for (int v = 0; v < 8; ++v) {
          int i = mbase + v;
          float hv = 0.f;
          if (i < K_REGION) hv = SG[i * N_KERNELS + nt] * p[m][v];
          u.h[i][n] = (__bf16)hv;
        }
      }
    }
  }
  __syncthreads();

  // ---- GEMM 3: h2 = h @ W1^T, tanh(+b1), dot w2, mean -------------------
  float sacc = 0.f;
  {
    v8f zero = {0.f, 0.f, 0.f, 0.f, 0.f, 0.f, 0.f, 0.f};
    for (int nt = 0; nt < 8; ++nt) {
      v8f hacc[3];
#pragma unroll
      for (int m = 0; m < 3; ++m) hacc[m] = zero;
#pragma unroll
      for (int ks = 0; ks < HID_DIM; ks += 32) {
        v16bf bw = load_b(w1n, HID_DIM, nt * 16, ks, lane);
#pragma unroll
        for (int m = 0; m < 3; ++m) {
          v16bf ah = load_a(&u.h[0][0], HID_DIM, 16 * m, ks, lane);
          hacc[m] = wmma_bf16(ah, bw, hacc[m]);
        }
      }
      int n = nt * 16 + (lane & 15);
      float bb = b1[n];
      float wv = w2n[n];
#pragma unroll
      for (int m = 0; m < 3; ++m) {
        int mbase = m * 16 + ((lane & 16) ? 8 : 0);
#pragma unroll
        for (int v = 0; v < 8; ++v) {
          int i = mbase + v;
          if (i < K_REGION) sacc += tanhf(hacc[m][v] + bb) * wv;
        }
      }
    }
  }
#pragma unroll
  for (int off = 16; off >= 1; off >>= 1) sacc += __shfl_xor(sacc, off, 32);
  if (lane == 0) out[(size_t)b * N_CAPTION + c] = sacc * (1.f / 36.f) + b2[0];
}

// ---------------------------------------------------------------------------
// Host launcher
// ---------------------------------------------------------------------------
extern "C" void kernel_launch(void* const* d_in, const int* /*in_sizes*/,
                              int /*n_in*/, void* d_out, int /*out_size*/,
                              void* d_ws, size_t /*ws_size*/,
                              hipStream_t stream) {
  const float* images   = (const float*)d_in[0];
  const float* captions = (const float*)d_in[1];
  const float* bbox     = (const float*)d_in[2];
  const float* conv_w   = (const float*)d_in[3];
  const float* mean_rho = (const float*)d_in[4];
  const float* prec_rho = (const float*)d_in[5];
  const float* mean_th  = (const float*)d_in[6];
  const float* prec_th  = (const float*)d_in[7];
  const float* o1v      = (const float*)d_in[8];
  const float* o1g      = (const float*)d_in[9];
  const float* o1b      = (const float*)d_in[10];
  const float* o2v      = (const float*)d_in[11];
  const float* o2g      = (const float*)d_in[12];
  const float* o2b      = (const float*)d_in[13];
  float* out = (float*)d_out;

  char* ws = (char*)d_ws;
  size_t off = 0;
  auto take = [&](size_t bytes) -> char* {
    char* p = ws + off;
    off += (bytes + 255) & ~(size_t)255;
    return p;
  };
  __bf16* img_bf  = (__bf16*)take((size_t)128 * 48 * 1024 * 2);
  __bf16* cap_bf  = (__bf16*)take((size_t)128 * 48 * 1024 * 2);
  __bf16* capT_bf = (__bf16*)take((size_t)128 * 1024 * 64 * 2);
  float*  qn      = (float*)take((size_t)128 * 36 * 32 * 4);
  float*  sg      = (float*)take((size_t)128 * 36 * 8 * 4);
  __bf16* convT   = (__bf16*)take((size_t)128 * 32 * 2);
  __bf16* w1n     = (__bf16*)take((size_t)128 * 128 * 2);
  float*  w2n     = (float*)take((size_t)128 * 4);

  cvt_img_kernel<<<(128 * 48 * 1024) / 256, 256, 0, stream>>>(images, img_bf);
  cvt_cap_kernel<<<(128 * 48 * 1024) / 256, 256, 0, stream>>>(captions, cap_bf);
  cvt_capT_kernel<<<(128 * 1024 * 64) / 256, 256, 0, stream>>>(captions,
                                                               capT_bf);
  qnorm_kernel<<<(128 * 36 * 32) / 256, 256, 0, stream>>>(images, qn);
  sgw_kernel<<<(128 * 36 + 255) / 256, 256, 0, stream>>>(
      bbox, mean_rho, prec_rho, mean_th, prec_th, sg);
  prep_weights_kernel<<<1, 128, 0, stream>>>(o1v, o1g, o2v, o2g, conv_w, w1n,
                                             w2n, convT);

  dim3 grid(N_IMAGE, N_CAPTION);
  pair_kernel<<<grid, 32, 0, stream>>>(images, img_bf, cap_bf, capT_bf, qn, sg,
                                       convT, w1n, o1b, w2n, o2b, out);
}
